// AttnDecoderRNN_71064528880216
// MI455X (gfx1250) — matrix operand order repaired and checked by reading, not compiled, not run on hardware
//
#include <hip/hip_runtime.h>
#include <hip/hip_bf16.h>

// ---------------------------------------------------------------------------
// AttnDecoderRNN on gfx1250 (MI455X): 48 sequential decoder steps.
// GEMMs: v_wmma_f32_16x16x32_f16, f16 operands (pre-converted, L2-resident),
// f32 accumulation. Tiles staged Global->LDS by the Tensor Data Mover
// (tensor_load_to_lds, TENSORcnt), double-buffered, overlapped with WMMA.
// ---------------------------------------------------------------------------

#define B_SZ   512
#define S_OUT  48
#define S_IN   96
#define D_OUT  256
#define D_HID  512

typedef _Float16 half16 __attribute__((ext_vector_type(16)));
typedef _Float16 half8  __attribute__((ext_vector_type(8)));
typedef float    float8 __attribute__((ext_vector_type(8)));
typedef unsigned int uint32x4 __attribute__((ext_vector_type(4)));
typedef int          int32x4  __attribute__((ext_vector_type(4)));
typedef int          int32x8  __attribute__((ext_vector_type(8)));

// ---------------- WMMA GEMM: C = act(A @ W^T + bias) -----------------------
// A: MxK f16 (row stride lda), W: NxK f16 (row stride ldw, Linear layout),
// bias: f32[N]. Output f32 (Cf) or f16 (Ch), row stride ldc.
// Block = 256 thr = 8 waves as 2(M) x 4(N); block tile 64 x 128; K step 32.
#define BM 64
#define BN 128
#define BK 32
#define LDS_PITCH 40   // halves per row (80B): TDM pads 16B after each 64B row

#if defined(__gfx1250__)
// Issue one TDM 2D tile load: rows x 32 halves from global (row stride lda
// elements) into LDS at byte offset lds_byte, padded to LDS_PITCH rows.
// clang-23 / amdgpu-toolchain builtin: 6 args
//   (uint32x4 g0, int32x8 g1, int32x4 g2, int32x4 g3, int32x8 gx, i32 cpol)
__device__ __forceinline__ void tdm_load_tile(const _Float16* g, long lda,
                                              unsigned lds_byte, int rows)
{
    unsigned long long ga = (unsigned long long)(uintptr_t)g;
    uint32x4 g0;
    g0[0] = 1u;                                   // count=1, user descriptor
    g0[1] = lds_byte;                             // lds_addr [63:32]
    g0[2] = (unsigned)(ga & 0xFFFFFFFFull);       // global_addr [95:64]
    g0[3] = (unsigned)((ga >> 32) & 0x01FFFFFFull) | (2u << 30); // [120:96], type=2

    int32x8 g1;
    g1[0] = (1 << 16)    // data_size = 2 bytes
          | (1 << 20)    // pad_enable
          | (3 << 22)    // pad_interval: pad after 16 DWORDs (64B row)
          | (3 << 25);   // pad_amount: 4 DWORDs (16B) -> pitch 40 halves
    g1[1] = (BK & 0xFFFF) << 16;                  // tensor_dim0[15:0]  = 32
    g1[2] = (rows & 0xFFFF) << 16;                // tensor_dim1[15:0]
    g1[3] = (BK & 0xFFFF) << 16;                  // tile_dim0 = 32
    g1[4] = rows & 0xFFFF;                        // tile_dim1 (tile_dim2 = 0)
    g1[5] = (int)(lda & 0xFFFFFFFFL);             // tensor_dim0_stride lo
    g1[6] = (int)((lda >> 32) & 0xFFFFL);         // stride hi (dim1_stride=0)
    g1[7] = 0;

    int32x4 z4 = {0, 0, 0, 0};                    // 2D: groups 2/3 unused
    int32x8 z8 = {0, 0, 0, 0, 0, 0, 0, 0};
    __builtin_amdgcn_tensor_load_to_lds(g0, g1, z4, z4, z8, 0);
}
#endif

__global__ __launch_bounds__(256)
void gemm_wmma_bias_act(const _Float16* __restrict__ A, int lda,
                        const _Float16* __restrict__ W, int ldw,
                        const float* __restrict__ bias,
                        float* __restrict__ Cf, _Float16* __restrict__ Ch,
                        int ldc, int K, int act)
{
#if defined(__gfx1250__)
    __shared__ _Float16 As[2][BM][LDS_PITCH];
    __shared__ _Float16 Bs[2][BN][LDS_PITCH];

    const int tid  = threadIdx.x;
    const int lane = tid & 31;
    const int wave = tid >> 5;
    const int m_base = (wave >> 2) * 32;   // 0,32
    const int n_base = (wave & 3)  * 32;   // 0,32,64,96
    const int blockM = blockIdx.x * BM;
    const int blockN = blockIdx.y * BN;

    const int l15  = lane & 15;
    const int hsel = lane >> 4;            // 0: lanes 0-15, 1: lanes 16-31

    const unsigned as_base = (unsigned)(uintptr_t)&As[0][0][0];
    const unsigned bs_base = (unsigned)(uintptr_t)&Bs[0][0][0];
    const unsigned as_sz = BM * LDS_PITCH * 2;   // bytes per A buffer
    const unsigned bs_sz = BN * LDS_PITCH * 2;   // bytes per B buffer
    const bool producer = (wave == 0);

    float8 acc[2][2] = {{{}, {}}, {{}, {}}};

    // prologue: DMA first tiles into buffer 0
    if (producer) {
        tdm_load_tile(&A[(long)blockM * lda], lda, as_base, BM);
        tdm_load_tile(&W[(long)blockN * ldw], ldw, bs_base, BN);
    }

    const int nk = K / BK;
    for (int ki = 0; ki < nk; ++ki) {
        const int buf = ki & 1;
        if (producer) {
            if (ki + 1 < nk) {   // DMA next tiles into the other buffer
                tdm_load_tile(&A[(long)blockM * lda + (ki + 1) * BK], lda,
                              as_base + (buf ^ 1) * as_sz, BM);
                tdm_load_tile(&W[(long)blockN * ldw + (ki + 1) * BK], ldw,
                              bs_base + (buf ^ 1) * bs_sz, BN);
                // in-order TDM: <=2 outstanding => current buffer complete
                __builtin_amdgcn_s_wait_tensorcnt(2);
            } else {
                __builtin_amdgcn_s_wait_tensorcnt(0);
            }
        }
        __syncthreads();   // release consumers: buf is ready

        // A fragments (ISA 16-bit A 16x32): lanes 0-15 K 0-7 & 16-23,
        // lanes 16-31 K 8-15 & 24-31.
        half16 af[2];
        #pragma unroll
        for (int mt = 0; mt < 2; ++mt) {
            half8 lo = *(const half8*)&As[buf][m_base + mt * 16 + l15][hsel * 8];
            half8 hi = *(const half8*)&As[buf][m_base + mt * 16 + l15][hsel * 8 + 16];
            #pragma unroll
            for (int i = 0; i < 8; ++i) { af[mt][i] = lo[i]; af[mt][8 + i] = hi[i]; }
        }
        // B fragments: lane n = l15; lanes 0-15 K 0-7 & 16-23, lanes 16-31 K 8-15 & 24-31.
        half16 bf[2];
        #pragma unroll
        for (int nt = 0; nt < 2; ++nt) {
            half8 lo = *(const half8*)&Bs[buf][n_base + nt * 16 + l15][hsel * 8];
            half8 hi = *(const half8*)&Bs[buf][n_base + nt * 16 + l15][hsel * 8 + 16];
            #pragma unroll
            for (int i = 0; i < 8; ++i) { bf[nt][i] = lo[i]; bf[nt][8 + i] = hi[i]; }
        }

        #pragma unroll
        for (int mt = 0; mt < 2; ++mt)
            #pragma unroll
            for (int nt = 0; nt < 2; ++nt)
                acc[mt][nt] = __builtin_amdgcn_wmma_f32_16x16x32_f16(
                    false, af[mt], false, bf[nt],
                    (short)0, acc[mt][nt], false, false);

        __syncthreads();   // all consumers done before buf is DMA-overwritten
    }

    // C/D layout: VGPR r -> M = r (+8 for lanes 16-31), N = l15.
    #pragma unroll
    for (int mt = 0; mt < 2; ++mt) {
        #pragma unroll
        for (int nt = 0; nt < 2; ++nt) {
            const int col = blockN + n_base + nt * 16 + l15;
            const float bv = bias[col];
            #pragma unroll
            for (int r = 0; r < 8; ++r) {
                int row = blockM + m_base + mt * 16 + r + hsel * 8;
                float v = acc[mt][nt][r] + bv;
                if (act == 1) v = v > 0.0f ? v : 0.0f;
                if (Ch) Ch[row * ldc + col] = (_Float16)v;
                else    Cf[row * ldc + col] = v;
            }
        }
    }
#endif
}

// -------- attention: logits -> softmax -> context; one block per batch -----
__global__ __launch_bounds__(256)
void attn_context_kernel(const float* __restrict__ target,   // (B,48,256)
                         const float* __restrict__ h1,       // (B,512) f32
                         const float* __restrict__ enc,      // (B,96,512)
                         const float* __restrict__ W_attn,   // (96,768)
                         const float* __restrict__ b_attn,   // (96)
                         float* __restrict__ attn_out,       // (B,48,96)
                         _Float16* __restrict__ cat_h,       // (B,768) f16
                         int t)
{
    __shared__ float xh[D_OUT + D_HID];   // 768
    __shared__ float aw[S_IN];            // 96
    __shared__ float red[2];

    const int b   = blockIdx.x;
    const int tid = threadIdx.x;

    float xv = target[(b * S_OUT + t) * D_OUT + tid];   // tid in [0,256)
    xh[tid] = xv;
    cat_h[b * 768 + tid] = (_Float16)xv;
    for (int i = tid; i < D_HID; i += 256) xh[D_OUT + i] = h1[b * D_HID + i];
    __syncthreads();

    if (tid < S_IN) {
        const float* wr = W_attn + tid * 768;
        float a = b_attn[tid];
        for (int k = 0; k < 768; ++k) a += xh[k] * wr[k];
        aw[tid] = a;
    }
    __syncthreads();

    if (tid == 0) {
        float m = aw[0];
        for (int s = 1; s < S_IN; ++s) m = fmaxf(m, aw[s]);
        float ssum = 0.0f;
        for (int s = 0; s < S_IN; ++s) ssum += __expf(aw[s] - m);
        red[0] = m;
        red[1] = 1.0f / ssum;
    }
    __syncthreads();

    if (tid < S_IN) {
        float e = __expf(aw[tid] - red[0]) * red[1];
        aw[tid] = e;
        attn_out[(b * S_OUT + t) * S_IN + tid] = e;
    }
    __syncthreads();

    const float* encb = enc + (long)b * S_IN * D_HID;
    for (int j = tid; j < D_HID; j += 256) {
        float a = 0.0f;
        for (int s = 0; s < S_IN; ++s) a += aw[s] * encb[s * D_HID + j];
        cat_h[b * 768 + D_OUT + j] = (_Float16)a;   // context into cat[:,256:]
    }
}

// ---------------- fused GRU gates, in-place hidden update ------------------
__global__ __launch_bounds__(256)
void gru_gate_kernel(const float* __restrict__ gi,  // (B,1536)
                     const float* __restrict__ gh,  // (B,1536)
                     float* __restrict__ h,         // (B,512) f32, updated
                     _Float16* __restrict__ h16)    // (B,512) f16 mirror
{
    int idx = blockIdx.x * blockDim.x + threadIdx.x;   // < B*512
    int b = idx >> 9;
    int j = idx & 511;
    const float* gib = gi + b * 3 * D_HID;
    const float* ghb = gh + b * 3 * D_HID;
    float r  = 1.0f / (1.0f + __expf(-(gib[j] + ghb[j])));
    float z  = 1.0f / (1.0f + __expf(-(gib[D_HID + j] + ghb[D_HID + j])));
    float nn = tanhf(gib[2 * D_HID + j] + r * ghb[2 * D_HID + j]);
    float hv = (1.0f - z) * nn + z * h[idx];
    h[idx]   = hv;
    h16[idx] = (_Float16)hv;
}

__global__ __launch_bounds__(256)
void copy_f32_kernel(const float* __restrict__ src, float* __restrict__ dst, int n)
{
    int i = blockIdx.x * blockDim.x + threadIdx.x;
    if (i < n) dst[i] = src[i];
}

__global__ __launch_bounds__(256)
void cvt_f32_to_f16_kernel(const float* __restrict__ src,
                           _Float16* __restrict__ dst, int n)
{
    int i = blockIdx.x * blockDim.x + threadIdx.x;
    if (i < n) dst[i] = (_Float16)src[i];
}

// ---------------------------------------------------------------------------
extern "C" void kernel_launch(void* const* d_in, const int* in_sizes, int n_in,
                              void* d_out, int out_size, void* d_ws, size_t ws_size,
                              hipStream_t stream) {
    (void)in_sizes; (void)n_in; (void)out_size; (void)ws_size;

    const float* target = (const float*)d_in[0];
    const float* hidden = (const float*)d_in[1];
    const float* enc    = (const float*)d_in[2];
    const float* W_attn = (const float*)d_in[3];
    const float* b_attn = (const float*)d_in[4];
    const float* W_comb = (const float*)d_in[5];
    const float* b_comb = (const float*)d_in[6];
    const float* Wih0   = (const float*)d_in[7];
    const float* Whh0   = (const float*)d_in[8];
    const float* bih0   = (const float*)d_in[9];
    const float* bhh0   = (const float*)d_in[10];
    const float* Wih1   = (const float*)d_in[11];
    const float* Whh1   = (const float*)d_in[12];
    const float* bih1   = (const float*)d_in[13];
    const float* bhh1   = (const float*)d_in[14];
    const float* Wo0    = (const float*)d_in[15];
    const float* bo0    = (const float*)d_in[16];
    const float* Wo1    = (const float*)d_in[17];
    const float* bo1    = (const float*)d_in[18];
    const float* Wo2    = (const float*)d_in[19];
    const float* bo2    = (const float*)d_in[20];

    // output layout: output (B,48,256) | hidden_f (2,B,512) | attn (B,48,96)
    float* out  = (float*)d_out;
    float* hid  = out + (long)B_SZ * S_OUT * D_OUT;
    float* hid0 = hid;
    float* hid1 = hid + (long)B_SZ * D_HID;
    float* attn = hid + 2L * B_SZ * D_HID;

    // ---- workspace layout ----
    float* ws = (float*)d_ws;
    float* gi = ws;                              // B*1536 f32
    float* gh = gi + (long)B_SZ * 3 * D_HID;     // B*1536 f32
    _Float16* hbase = (_Float16*)(gh + (long)B_SZ * 3 * D_HID);
    _Float16* cat_h  = hbase;                    // B*768
    _Float16* x_h    = cat_h  + (long)B_SZ * 768;      // B*256
    _Float16* o1_h   = x_h    + (long)B_SZ * D_OUT;    // B*512
    _Float16* o2_h   = o1_h   + (long)B_SZ * D_HID;    // B*256
    _Float16* h0_h   = o2_h   + (long)B_SZ * D_OUT;    // B*512
    _Float16* h1_h   = h0_h   + (long)B_SZ * D_HID;    // B*512
    _Float16* Wc_h   = h1_h   + (long)B_SZ * D_HID;    // 256*768
    _Float16* Wih0_h = Wc_h   + 256L * 768;            // 1536*256
    _Float16* Whh0_h = Wih0_h + 1536L * 256;           // 1536*512
    _Float16* Wih1_h = Whh0_h + 1536L * 512;           // 1536*512
    _Float16* Whh1_h = Wih1_h + 1536L * 512;           // 1536*512
    _Float16* Wo0_h  = Whh1_h + 1536L * 512;           // 512*512
    _Float16* Wo1_h  = Wo0_h  + 512L * 512;            // 256*512
    _Float16* Wo2_h  = Wo1_h  + 256L * 512;            // 256*256

    const dim3 blk(256);
    #define CVT(src, dst, n) \
        cvt_f32_to_f16_kernel<<<((n) + 255) / 256, blk, 0, stream>>>(src, dst, n)

    // once per call: working hidden state + f16 weight/hidden mirrors
    {
        int n = 2 * B_SZ * D_HID;
        copy_f32_kernel<<<(n + 255) / 256, blk, 0, stream>>>(hidden, hid, n);
    }
    CVT(hidden,                 h0_h,   B_SZ * D_HID);
    CVT(hidden + B_SZ * D_HID,  h1_h,   B_SZ * D_HID);
    CVT(W_comb, Wc_h,   256 * 768);
    CVT(Wih0,   Wih0_h, 1536 * 256);
    CVT(Whh0,   Whh0_h, 1536 * 512);
    CVT(Wih1,   Wih1_h, 1536 * 512);
    CVT(Whh1,   Whh1_h, 1536 * 512);
    CVT(Wo0,    Wo0_h,  512 * 512);
    CVT(Wo1,    Wo1_h,  256 * 512);
    CVT(Wo2,    Wo2_h,  256 * 256);

    const dim3 g_256 (B_SZ / BM, D_OUT / BN);          // 8 x 2
    const dim3 g_512 (B_SZ / BM, D_HID / BN);          // 8 x 4
    const dim3 g_1536(B_SZ / BM, (3 * D_HID) / BN);    // 8 x 12

    for (int t = 0; t < S_OUT; ++t) {
        // 1) attention + context -> cat_h = [x_t | weighted]  (f16)
        attn_context_kernel<<<B_SZ, blk, 0, stream>>>(
            target, hid1, enc, W_attn, b_attn, attn, cat_h, t);

        // 2) x = relu(cat @ W_comb^T + b_comb)   -> f16
        gemm_wmma_bias_act<<<g_256, blk, 0, stream>>>(
            cat_h, 768, Wc_h, 768, b_comb, nullptr, x_h, D_OUT, 768, 1);

        // 3) GRU layer 0
        gemm_wmma_bias_act<<<g_1536, blk, 0, stream>>>(
            x_h, D_OUT, Wih0_h, D_OUT, bih0, gi, nullptr, 3 * D_HID, D_OUT, 0);
        gemm_wmma_bias_act<<<g_1536, blk, 0, stream>>>(
            h0_h, D_HID, Whh0_h, D_HID, bhh0, gh, nullptr, 3 * D_HID, D_HID, 0);
        gru_gate_kernel<<<(B_SZ * D_HID) / 256, blk, 0, stream>>>(gi, gh, hid0, h0_h);

        // 4) GRU layer 1
        gemm_wmma_bias_act<<<g_1536, blk, 0, stream>>>(
            h0_h, D_HID, Wih1_h, D_HID, bih1, gi, nullptr, 3 * D_HID, D_HID, 0);
        gemm_wmma_bias_act<<<g_1536, blk, 0, stream>>>(
            h1_h, D_HID, Whh1_h, D_HID, bhh1, gh, nullptr, 3 * D_HID, D_HID, 0);
        gru_gate_kernel<<<(B_SZ * D_HID) / 256, blk, 0, stream>>>(gi, gh, hid1, h1_h);

        // 5) output MLP; final GEMM writes f32 strided into out[:, t, :]
        gemm_wmma_bias_act<<<g_512, blk, 0, stream>>>(
            h1_h, D_HID, Wo0_h, D_HID, bo0, nullptr, o1_h, D_HID, D_HID, 1);
        gemm_wmma_bias_act<<<g_256, blk, 0, stream>>>(
            o1_h, D_HID, Wo1_h, D_HID, bo1, nullptr, o2_h, D_OUT, D_HID, 1);
        gemm_wmma_bias_act<<<g_256, blk, 0, stream>>>(
            o2_h, D_OUT, Wo2_h, D_OUT, bo2, out + (long)t * D_OUT, nullptr,
            S_OUT * D_OUT, D_OUT, 0);
    }
    #undef CVT
}
